// ProbSparseSelfAttention_70205535420418
// MI455X (gfx1250) — matrix operand order repaired and checked by reading, not compiled, not run on hardware
//
#include <hip/hip_runtime.h>
#include <hip/hip_bf16.h>

// ProbSparse self-attention for MI455X (gfx1250, wave32, WMMA).
// GEMMs via v_wmma_f32_16x16x32_bf16 (f32 accumulate).
// Data movement: async global->LDS (ASYNCcnt-tracked, verified present on this
// toolchain), double-buffered k-slab staging in the GEMM, 128-key slab staging
// in attention.

#define D_MODEL 512
#define NHEAD   8
#define DHEAD   64
#define SEQ_L   2048
#define BATCH   4
#define MROWS   (BATCH * SEQ_L)   // 8192
#define TOPK_U  38                // floor(5 * ln(2048))

typedef __attribute__((ext_vector_type(16))) __bf16 v16bf;
typedef __attribute__((ext_vector_type(8)))  float  v8f;
typedef __attribute__((ext_vector_type(4)))  int    v4i;

union BF16Frag { v16bf v; unsigned short u[16]; };

// ---- CDNA5 async global->LDS path (probe via __has_builtin) ----------------
#if defined(__has_builtin)
#  if __has_builtin(__builtin_amdgcn_global_load_async_to_lds_b128) && \
      __has_builtin(__builtin_amdgcn_s_wait_asynccnt)
#    define USE_ASYNC_LDS 1
#  endif
#endif
#ifndef USE_ASYNC_LDS
#  define USE_ASYNC_LDS 0
#endif

typedef __attribute__((address_space(1))) v4i gv4i;   // global
typedef __attribute__((address_space(3))) v4i lv4i;   // LDS

__device__ inline void async_copy16(const void* g, void* l) {
#if USE_ASYNC_LDS
  __builtin_amdgcn_global_load_async_to_lds_b128((gv4i*)g, (lv4i*)l, 0, 0);
#else
  *(uint4*)l = *(const uint4*)g;
#endif
}
__device__ inline void async_wait() {
#if USE_ASYNC_LDS
  __builtin_amdgcn_s_wait_asynccnt(0);
#endif
}

__device__ inline unsigned short f32_to_bf16_bits(float f) {
  unsigned u = __float_as_uint(f);
  unsigned r = u + 0x7FFFu + ((u >> 16) & 1u);   // round-to-nearest-even
  return (unsigned short)(r >> 16);
}

__device__ inline v8f wmma_bf16(const BF16Frag& a, const BF16Frag& b, v8f c) {
  return __builtin_amdgcn_wmma_f32_16x16x32_bf16(false, a.v, false, b.v,
                                                 (short)0, c, false, false);
}

// monotonic float -> uint key (larger float => larger key)
__device__ inline unsigned f32_key(float f) {
  unsigned u = __float_as_uint(f);
  return (u & 0x80000000u) ? ~u : (u | 0x80000000u);
}

__device__ inline float wave_max(float v) {
  #pragma unroll
  for (int m = 16; m; m >>= 1) v = fmaxf(v, __shfl_xor(v, m, 32));
  return v;
}
__device__ inline float wave_sum(float v) {
  #pragma unroll
  for (int m = 16; m; m >>= 1) v += __shfl_xor(v, m, 32);
  return v;
}
__device__ inline unsigned wave_sum_u(unsigned v) {
  #pragma unroll
  for (int m = 16; m; m >>= 1) v += (unsigned)__shfl_xor((int)v, m, 32);
  return v;
}

// ---------------------------------------------------------------------------
// fp32 -> bf16 bits conversion
// ---------------------------------------------------------------------------
__global__ void cvt_bf16_kernel(const float* __restrict__ in,
                                unsigned short* __restrict__ out, int n) {
  int i = blockIdx.x * blockDim.x + threadIdx.x;
  if (i < n) out[i] = f32_to_bf16_bits(in[i]);
}

// ---------------------------------------------------------------------------
// Tiled bf16 GEMM: C[M=8192, N=512] = A[M,512] * W[512,512] + bias
// mode 0: scatter bf16 output into [B, H, L, D] head-major layout (QKV)
// mode 1: plain f32 output row-major (final projection into d_out)
// 256 threads = 8 waves; 64x64 tile, k-step 32; double-buffered LDS staging.
// ---------------------------------------------------------------------------
__global__ __launch_bounds__(256)
void gemm_bf16_kernel(const unsigned short* __restrict__ A,
                      const unsigned short* __restrict__ W,
                      const float* __restrict__ bias,
                      unsigned short* __restrict__ outQKV,
                      float* __restrict__ outF32,
                      int mode)
{
  __shared__ unsigned short As[2][64][32];   // 2 x 4 KB
  __shared__ unsigned short Bs[2][32][64];   // 2 x 4 KB
  const int tid  = threadIdx.x;
  const int lane = tid & 31, wave = tid >> 5;
  const int mBase = blockIdx.x * 64;
  const int nBase = blockIdx.y * 64;
  const int mOff  = (wave & 3) * 16;
  const int nOff  = (wave >> 2) * 32;
  const int half  = lane >> 4;

  v8f c0 = {}; v8f c1 = {};

  const int ar = tid >> 2, ac = (tid & 3) * 8;   // A stage: 64 rows x 4 chunks
  const int br = tid >> 3, bc = (tid & 7) * 8;   // B stage: 32 rows x 8 chunks

  // prologue: stage k-slab 0 into buffer 0
  async_copy16(&A[(size_t)(mBase + ar) * D_MODEL + ac], &As[0][ar][ac]);
  async_copy16(&W[(size_t)br * D_MODEL + nBase + bc],   &Bs[0][br][bc]);
  async_wait();
  __syncthreads();

  for (int k0 = 0; k0 < D_MODEL; k0 += 32) {
    const int cur = (k0 >> 5) & 1;
    // stream next slab into the other buffer while we compute on this one
    if (k0 + 32 < D_MODEL) {
      async_copy16(&A[(size_t)(mBase + ar) * D_MODEL + k0 + 32 + ac],
                   &As[cur ^ 1][ar][ac]);
      async_copy16(&W[(size_t)(k0 + 32 + br) * D_MODEL + nBase + bc],
                   &Bs[cur ^ 1][br][bc]);
      __builtin_prefetch(&A[(size_t)(mBase + ar) * D_MODEL + k0 + 64 + ac], 0, 1);
      __builtin_prefetch(&W[(size_t)(k0 + 64 + br) * D_MODEL + nBase + bc], 0, 1);
    }

    // A fragment (16x32, ISA 7.12.2 layout)
    BF16Frag a, b0, b1;
    const int mr = mOff + (lane & 15);
    #pragma unroll
    for (int v = 0; v < 8; ++v) {
      int kb = ((v < 4) ? 0 : 16) + half * 8 + (v & 3) * 2;
      a.u[2*v]   = As[cur][mr][kb];
      a.u[2*v+1] = As[cur][mr][kb + 1];
    }
    const int bn = lane & 15;
    #pragma unroll
    for (int v = 0; v < 8; ++v) {
      int kb = ((v < 4) ? 0 : 16) + half * 8 + (v & 3) * 2;
      b0.u[2*v]   = Bs[cur][kb][nOff + bn];
      b0.u[2*v+1] = Bs[cur][kb + 1][nOff + bn];
      b1.u[2*v]   = Bs[cur][kb][nOff + 16 + bn];
      b1.u[2*v+1] = Bs[cur][kb + 1][nOff + 16 + bn];
    }
    c0 = wmma_bf16(a, b0, c0);
    c1 = wmma_bf16(a, b1, c1);

    async_wait();        // next-slab copies done (ASYNCcnt==0)
    __syncthreads();     // all waves' copies visible before buffer swap
  }

  // Epilogue: C/D layout = N across lanes 0-15, M across VGPRs (+8 hi half)
  const int nc = lane & 15;
  #pragma unroll
  for (int v = 0; v < 8; ++v) {
    int row = mBase + mOff + v + 8 * half;
    int col0 = nBase + nOff + nc;
    int col1 = col0 + 16;
    float f0 = c0[v] + bias[col0];
    float f1 = c1[v] + bias[col1];
    if (mode == 0) {
      int b = row >> 11, l = row & (SEQ_L - 1);
      int h0 = col0 >> 6, d0 = col0 & 63;
      int h1 = col1 >> 6, d1 = col1 & 63;
      outQKV[((size_t)(b * NHEAD + h0) * SEQ_L + l) * DHEAD + d0] = f32_to_bf16_bits(f0);
      outQKV[((size_t)(b * NHEAD + h1) * SEQ_L + l) * DHEAD + d1] = f32_to_bf16_bits(f1);
    } else {
      outF32[(size_t)row * D_MODEL + col0] = f0;
      outF32[(size_t)row * D_MODEL + col1] = f1;
    }
  }
}

// ---------------------------------------------------------------------------
// ProbSparse attention for one (b,h) and a tile of 16 queries.
// LDS: 16x2048 f32 score strip (128 KB) + 128x64 bf16 K/V slab stage (16 KB).
// Phase 1: S = QK^T/8 via WMMA over block-staged 128-key slabs.
// Phase 2: exact top-U=38 radix-select threshold + masked softmax (per wave).
// Phase 3: ctx = P @ V via WMMA over block-staged V slabs.
// ---------------------------------------------------------------------------
__global__ __launch_bounds__(256)
void attn_kernel(const unsigned short* __restrict__ Qb,
                 const unsigned short* __restrict__ Kb,
                 const unsigned short* __restrict__ Vb,
                 unsigned short* __restrict__ ctxb)
{
  extern __shared__ char smem[];
  float* S = (float*)smem;                                       // [16][SEQ_L]
  unsigned short* KV = (unsigned short*)(smem + 16 * SEQ_L * 4); // [128][64]

  const int bh    = blockIdx.y;
  const int qBase = blockIdx.x * 16;
  const unsigned short* Qh = Qb + (size_t)bh * SEQ_L * DHEAD;
  const unsigned short* Kh = Kb + (size_t)bh * SEQ_L * DHEAD;
  const unsigned short* Vh = Vb + (size_t)bh * SEQ_L * DHEAD;
  const int tid = threadIdx.x, lane = tid & 31, wave = tid >> 5;
  const int half = lane >> 4;

  // Cache the two Q fragments (D=64 -> 2 k-chunks of 32) in registers.
  BF16Frag aq[2];
  {
    const int m = lane & 15;
    #pragma unroll
    for (int ch = 0; ch < 2; ++ch)
      #pragma unroll
      for (int v = 0; v < 8; ++v) {
        int kb = ch * 32 + ((v < 4) ? 0 : 16) + half * 8 + (v & 3) * 2;
        aq[ch].u[2*v]   = Qh[(size_t)(qBase + m) * DHEAD + kb];
        aq[ch].u[2*v+1] = Qh[(size_t)(qBase + m) * DHEAD + kb + 1];
      }
  }

  // -------- Phase 1: scores over 16 slabs of 128 keys ----------------------
  for (int slab = 0; slab < SEQ_L / 128; ++slab) {
    const unsigned short* src = Kh + (size_t)slab * 128 * DHEAD;
    #pragma unroll
    for (int j = 0; j < 4; ++j) {        // 1024 uint4 chunks / 256 threads
      int c = tid + j * 256;
      async_copy16(&src[c * 8], &KV[c * 8]);
    }
    async_wait();
    __syncthreads();

    const int keyBase = wave * 16;       // this wave's 16 keys within the slab
    v8f acc = {};
    #pragma unroll
    for (int ch = 0; ch < 2; ++ch) {
      BF16Frag b;
      const int key = keyBase + (lane & 15);
      #pragma unroll
      for (int v = 0; v < 8; ++v) {
        int kb = ch * 32 + ((v < 4) ? 0 : 16) + half * 8 + (v & 3) * 2;
        b.u[2*v]   = KV[key * DHEAD + kb];
        b.u[2*v+1] = KV[key * DHEAD + kb + 1];
      }
      acc = wmma_bf16(aq[ch], b, acc);
    }
    #pragma unroll
    for (int v = 0; v < 8; ++v) {
      int row = v + 8 * half;
      int col = slab * 128 + keyBase + (lane & 15);
      S[row * SEQ_L + col] = acc[v] * 0.125f;   // 1/sqrt(64)
    }
    __syncthreads();
  }

  // -------- Phase 2: exact U-th-largest threshold + masked softmax ---------
  for (int row = wave; row < 16; row += 8) {
    float* Srow = S + row * SEQ_L;
    float mx = -3.4e38f;
    for (int i = lane; i < SEQ_L; i += 32) mx = fmaxf(mx, Srow[i]);
    mx = wave_max(mx);

    unsigned thr = 0;
    for (int bit = 31; bit >= 0; --bit) {
      unsigned cand = thr | (1u << bit);
      unsigned cnt = 0;
      for (int i = lane; i < SEQ_L; i += 32)
        cnt += (f32_key(Srow[i]) >= cand) ? 1u : 0u;
      cnt = wave_sum_u(cnt);
      if (cnt >= (unsigned)TOPK_U) thr = cand;
    }

    float sum = 0.f;
    for (int i = lane; i < SEQ_L; i += 32) {
      float s = Srow[i];
      float p = (f32_key(s) >= thr) ? __expf(s - mx) : 0.f;
      Srow[i] = p;
      sum += p;
    }
    sum = wave_sum(sum);
    const float inv = 1.f / sum;
    for (int i = lane; i < SEQ_L; i += 32) Srow[i] *= inv;
  }
  __syncthreads();

  // -------- Phase 3: ctx = P @ V over 16 V-slabs ---------------------------
  v8f acc = {};
  const int nOff = (wave & 3) * 16;
  for (int slab = 0; slab < SEQ_L / 128; ++slab) {
    const unsigned short* src = Vh + (size_t)slab * 128 * DHEAD;
    #pragma unroll
    for (int j = 0; j < 4; ++j) {
      int c = tid + j * 256;
      async_copy16(&src[c * 8], &KV[c * 8]);
    }
    async_wait();
    __syncthreads();

    if (wave < 4) {
      const int m = lane & 15;
      const int kr = (lane & 15) + 16 * half;
      #pragma unroll
      for (int j = 0; j < 4; ++j) {      // 4 x 32-key WMMA steps per slab
        const int kkL = j * 32;
        const int kkG = slab * 128 + kkL;
        BF16Frag a, b;
        #pragma unroll
        for (int v = 0; v < 8; ++v) {
          int kb = ((v < 4) ? 0 : 16) + half * 8 + (v & 3) * 2;
          a.u[2*v]   = f32_to_bf16_bits(S[m * SEQ_L + kkG + kb]);
          a.u[2*v+1] = f32_to_bf16_bits(S[m * SEQ_L + kkG + kb + 1]);
        }
        #pragma unroll
        for (int v = 0; v < 8; ++v) {
          int kb = ((v < 4) ? 0 : 16) + half * 8 + (v & 3) * 2;
          b.u[2*v]   = KV[(kkL + kb) * DHEAD + nOff + m];
          b.u[2*v+1] = KV[(kkL + kb + 1) * DHEAD + nOff + m];
        }
        acc = wmma_bf16(a, b, acc);
      }
    }
    __syncthreads();
  }

  if (wave < 4) {
    const int bI = bh >> 3, h = bh & 7;
    #pragma unroll
    for (int v = 0; v < 8; ++v) {
      int row = v + 8 * half;
      int col = nOff + (lane & 15);
      ctxb[((size_t)(bI * SEQ_L) + qBase + row) * D_MODEL + h * DHEAD + col] =
          f32_to_bf16_bits(acc[v]);
    }
  }
}

// ---------------------------------------------------------------------------
extern "C" void kernel_launch(void* const* d_in, const int* in_sizes, int n_in,
                              void* d_out, int out_size, void* d_ws, size_t ws_size,
                              hipStream_t stream) {
  const float* x  = (const float*)d_in[0];
  const float* Wq = (const float*)d_in[1];
  const float* bq = (const float*)d_in[2];
  const float* Wk = (const float*)d_in[3];
  const float* bk = (const float*)d_in[4];
  const float* Wv = (const float*)d_in[5];
  const float* bv = (const float*)d_in[6];
  const float* Wo = (const float*)d_in[7];
  const float* bo = (const float*)d_in[8];
  float* out = (float*)d_out;

  // Workspace layout (bf16 bits, ~44 MB total)
  unsigned short* ws = (unsigned short*)d_ws;
  const size_t Mx = (size_t)MROWS * D_MODEL;
  const size_t Wn = (size_t)D_MODEL * D_MODEL;
  unsigned short* xb  = ws;
  unsigned short* wqb = xb  + Mx;
  unsigned short* wkb = wqb + Wn;
  unsigned short* wvb = wkb + Wn;
  unsigned short* wob = wvb + Wn;
  unsigned short* qb  = wob + Wn;
  unsigned short* kb  = qb  + Mx;
  unsigned short* vb  = kb  + Mx;
  unsigned short* cb  = vb  + Mx;

  cvt_bf16_kernel<<<(unsigned)((Mx + 255) / 256), 256, 0, stream>>>(x,  xb,  (int)Mx);
  cvt_bf16_kernel<<<(unsigned)((Wn + 255) / 256), 256, 0, stream>>>(Wq, wqb, (int)Wn);
  cvt_bf16_kernel<<<(unsigned)((Wn + 255) / 256), 256, 0, stream>>>(Wk, wkb, (int)Wn);
  cvt_bf16_kernel<<<(unsigned)((Wn + 255) / 256), 256, 0, stream>>>(Wv, wvb, (int)Wn);
  cvt_bf16_kernel<<<(unsigned)((Wn + 255) / 256), 256, 0, stream>>>(Wo, wob, (int)Wn);

  dim3 gGemm(MROWS / 64, D_MODEL / 64);   // 128 x 8
  dim3 blk(256);
  gemm_bf16_kernel<<<gGemm, blk, 0, stream>>>(xb, wqb, bq, qb, nullptr, 0);
  gemm_bf16_kernel<<<gGemm, blk, 0, stream>>>(xb, wkb, bk, kb, nullptr, 0);
  gemm_bf16_kernel<<<gGemm, blk, 0, stream>>>(xb, wvb, bv, vb, nullptr, 0);

  // 128 query tiles x 32 (b,h); 144 KB dynamic LDS (score strip + slab stage)
  attn_kernel<<<dim3(SEQ_L / 16, BATCH * NHEAD), blk,
                16 * SEQ_L * sizeof(float) + 128 * DHEAD * sizeof(unsigned short),
                stream>>>(qb, kb, vb, cb);

  gemm_bf16_kernel<<<gGemm, blk, 0, stream>>>(cb, wob, bo, nullptr, out, 1);
}
